// NeuralMem_88407606821030
// MI455X (gfx1250) — compile-verified, matrix operand order checked by reading, not run on hardware
//
#include <hip/hip_runtime.h>

#define HH 512
#define WW 512
#define CC 3
#define KK 32
#define SS 4
#define PP 10
#define NPATCH 126
#define NP (NPATCH * NPATCH)   // 15876
#define DD 3072
#define NMEM 4096
#define MPAD 15936             // 64 * 249, >= NP, multiple of 64
#define THRESH 0.5f

typedef __attribute__((ext_vector_type(16))) _Float16 v16h;
typedef __attribute__((ext_vector_type(8)))  _Float16 v8h;
typedef __attribute__((ext_vector_type(8)))  float    v8f;

union Frag { v16h v; v8h h[2]; };

// 16-bit A/B fragment (16x32 / 32x16): lane l<16 holds row (base + l), halves
// K=[0..7] and K=[16..23]; lane l>=16 holds row (base + l-16), halves K=[8..15]
// and K=[24..31].  Two 16-byte loads per lane from a row-major f16 row.
__device__ __forceinline__ v16h load_frag(const _Float16* rowk, int hiOff) {
    Frag f;
    f.h[0] = *(const v8h*)(rowk + hiOff);
    f.h[1] = *(const v8h*)(rowk + 16 + hiOff);
    return f.v;
}

// ---------------------------------------------------------------- pack mem
__global__ void pack_mem_kernel(const float* __restrict__ mem,
                                _Float16* __restrict__ memh,
                                float* __restrict__ mnorm2) {
    const int row = blockIdx.x;
    const float* src = mem + (size_t)row * DD;
    _Float16* dst = memh + (size_t)row * DD;
    float ssq = 0.f;
    for (int t = threadIdx.x; t < DD; t += 256) {
        float v = src[t];
        dst[t] = (_Float16)v;
        ssq += v * v;
    }
    __shared__ float red[256];
    red[threadIdx.x] = ssq;
    __syncthreads();
    for (int s = 128; s > 0; s >>= 1) {
        if (threadIdx.x < s) red[threadIdx.x] += red[threadIdx.x + s];
        __syncthreads();
    }
    if (threadIdx.x == 0) mnorm2[row] = red[0];
}

// ---------------------------------------------------------------- unfold u
__global__ void pack_u_kernel(const float* __restrict__ image,
                              _Float16* __restrict__ uh,
                              float* __restrict__ unorm2) {
    const int p = blockIdx.x;
    _Float16* dst = uh + (size_t)p * DD;
    if (p >= NP) {  // zero padding rows (whole block takes this branch)
        for (int t = threadIdx.x; t < DD; t += 256) dst[t] = (_Float16)0.f;
        if (threadIdx.x == 0) unorm2[p] = 0.f;
        return;
    }
    const int pr = p / NPATCH, pc = p % NPATCH;
    const int bh = pr * SS - PP, bw = pc * SS - PP;
    float ssq = 0.f;
    for (int t = threadIdx.x; t < DD; t += 256) {
        int c   = t >> 10;        // d = c*1024 + kr*32 + kc
        int rem = t & 1023;
        int kr  = rem >> 5;
        int kc  = rem & 31;
        int h = bh + kr, w = bw + kc;
        float v = (h >= 0 && h < HH && w >= 0 && w < WW)
                      ? image[(h * WW + w) * CC + c] : 0.f;
        dst[t] = (_Float16)v;
        ssq += v * v;
    }
    __shared__ float red[256];
    red[threadIdx.x] = ssq;
    __syncthreads();
    for (int s = 128; s > 0; s >>= 1) {
        if (threadIdx.x < s) red[threadIdx.x] += red[threadIdx.x + s];
        __syncthreads();
    }
    if (threadIdx.x == 0) unorm2[p] = red[0];
}

// ---------------------------------------------------------------- init
__global__ void init_kernel(unsigned long long* __restrict__ keys,
                            float* __restrict__ gmax) {
    int i = blockIdx.x * 256 + threadIdx.x;
    if (i < MPAD) keys[i] = 0xFFFFFFFFFFFFFFFFull;
    if (i == 0) *gmax = -3.4e38f;
}

// ------------------------------------------------- GEMM (WMMA f16) + argmin
__global__ void __launch_bounds__(256)
gemm_argmin_kernel(const _Float16* __restrict__ uh,
                   const _Float16* __restrict__ memh,
                   const float* __restrict__ unorm2,
                   const float* __restrict__ mnorm2,
                   unsigned long long* __restrict__ keys) {
    const int wave    = threadIdx.x >> 5;
    const int lane    = threadIdx.x & 31;
    const int laneLow = lane & 15;
    const int hiOff   = (lane >> 4) * 8;

    const int tile    = blockIdx.x * 8 + wave;       // one 64x64 tile per wave
    const int tiles_n = NMEM / 64;                   // 64
    const int m0 = (tile / tiles_n) * 64;
    const int n0 = (tile % tiles_n) * 64;

    v8f acc[4][4];
    v8f z = {};
#pragma unroll
    for (int i = 0; i < 4; i++)
#pragma unroll
        for (int j = 0; j < 4; j++) acc[i][j] = z;

    const _Float16* aBase = uh   + (size_t)(m0 + laneLow) * DD;
    const _Float16* bBase = memh + (size_t)(n0 + laneLow) * DD;

    for (int kk = 0; kk < DD; kk += 32) {
        v16h a[4], b[4];
#pragma unroll
        for (int i = 0; i < 4; i++)
            a[i] = load_frag(aBase + (size_t)(16 * i) * DD + kk, hiOff);
#pragma unroll
        for (int j = 0; j < 4; j++)
            b[j] = load_frag(bBase + (size_t)(16 * j) * DD + kk, hiOff);
#pragma unroll
        for (int i = 0; i < 4; i++)
#pragma unroll
            for (int j = 0; j < 4; j++)
                acc[i][j] = __builtin_amdgcn_wmma_f32_16x16x32_f16(
                    false, a[i], false, b[j], (short)0, acc[i][j],
                    false, false);
    }

    float mn[4];
#pragma unroll
    for (int j = 0; j < 4; j++) mn[j] = mnorm2[n0 + 16 * j + laneLow];

#pragma unroll
    for (int i = 0; i < 4; i++) {
#pragma unroll
        for (int r = 0; r < 8; r++) {
            const int row = m0 + 16 * i + r + hiOff;  // C/D layout: VGPR r, hi half +8
            const float un = unorm2[row];
            float best = 3.4e38f;
            unsigned bcol = 0u;
#pragma unroll
            for (int j = 0; j < 4; j++) {
                float d2 = un + mn[j] - 2.0f * acc[i][j][r];
                unsigned col = (unsigned)(n0 + 16 * j + laneLow);
                if (d2 < best || (d2 == best && col < bcol)) { best = d2; bcol = col; }
            }
            // min-reduce across the 16 lanes holding this row's columns
#pragma unroll
            for (int off = 1; off < 16; off <<= 1) {
                float    ov = __shfl_xor(best, off, 32);
                unsigned oc = __shfl_xor(bcol, off, 32);
                if (ov < best || (ov == best && oc < bcol)) { best = ov; bcol = oc; }
            }
            if (laneLow == 0) {
                float cl = best < 0.f ? 0.f : best;   // d2 >= 0; guard f16 rounding
                unsigned long long key =
                    ((unsigned long long)__float_as_uint(cl) << 32) |
                    (unsigned long long)bcol;
                atomicMin(&keys[row], key);
            }
        }
    }
}

// ---------------------------------------------------------------- select
__global__ void select_kernel(const unsigned long long* __restrict__ keys,
                              const long long* __restrict__ mapping,
                              int* __restrict__ sel) {
    int p = blockIdx.x * 256 + threadIdx.x;
    if (p >= NP) return;
    unsigned long long k = keys[p];
    float dmin   = __uint_as_float((unsigned)(k >> 32));
    unsigned idx = (unsigned)(k & 0xFFFFFFFFu);
    sel[p] = (dmin >= THRESH) ? -1 : (int)mapping[idx];
}

// ------------------------------------------------- fold (gather) + max
__global__ void fold_kernel(const float* __restrict__ image,
                            const float* __restrict__ mem2,
                            const int* __restrict__ sel,
                            float* __restrict__ canvas,
                            float* __restrict__ gmax) {
    int i = blockIdx.x * 256 + threadIdx.x;        // HWC flat index
    float acc = 0.f;
    if (i < HH * WW * CC) {
        int c = i % CC;
        int w = (i / CC) % WW;
        int h = i / (WW * CC);
        int hp = h + PP, wp = w + PP;              // canvas coords
        int pr_lo = (hp >= KK - 1) ? (hp - KK + SS) / SS : 0;
        int pr_hi = min(hp / SS, NPATCH - 1);
        int pc_lo = (wp >= KK - 1) ? (wp - KK + SS) / SS : 0;
        int pc_hi = min(wp / SS, NPATCH - 1);
        float img = image[i];
        for (int pr = pr_lo; pr <= pr_hi; ++pr) {
            int kr = hp - SS * pr;
            for (int pc = pc_lo; pc <= pc_hi; ++pc) {
                int kc = wp - SS * pc;
                int s = sel[pr * NPATCH + pc];
                float v = (s < 0) ? img
                                  : mem2[(size_t)s * DD + (c << 10) + kr * KK + kc];
                acc += v;
            }
        }
        canvas[i] = acc;
    } else {
        acc = -3.4e38f;
    }
    __shared__ float red[256];
    red[threadIdx.x] = acc;
    __syncthreads();
    for (int s = 128; s > 0; s >>= 1) {
        if (threadIdx.x < s)
            red[threadIdx.x] = fmaxf(red[threadIdx.x], red[threadIdx.x + s]);
        __syncthreads();
    }
    if (threadIdx.x == 0) atomicMax(gmax, red[0]);
}

// ---------------------------------------------------------------- normalize
__global__ void normalize_kernel(const float* __restrict__ canvas,
                                 const float* __restrict__ gmax,
                                 float* __restrict__ out) {
    int i = blockIdx.x * 256 + threadIdx.x;
    if (i < HH * WW * CC) out[i] = canvas[i] / (*gmax);
}

// ---------------------------------------------------------------- launch
extern "C" void kernel_launch(void* const* d_in, const int* in_sizes, int n_in,
                              void* d_out, int out_size, void* d_ws, size_t ws_size,
                              hipStream_t stream) {
    const float*     image   = (const float*)d_in[0];
    const float*     mem     = (const float*)d_in[1];
    const float*     mem2    = (const float*)d_in[2];
    const long long* mapping = (const long long*)d_in[3];
    float* out = (float*)d_out;

    char* ws = (char*)d_ws;
    size_t off = 0;
    auto alloc = [&](size_t bytes) -> void* {
        void* p = ws + off;
        off += (bytes + 255) & ~(size_t)255;
        return p;
    };
    _Float16*           uh     = (_Float16*)alloc((size_t)MPAD * DD * sizeof(_Float16));
    _Float16*           memh   = (_Float16*)alloc((size_t)NMEM * DD * sizeof(_Float16));
    float*              unorm2 = (float*)alloc((size_t)MPAD * sizeof(float));
    float*              mnorm2 = (float*)alloc((size_t)NMEM * sizeof(float));
    unsigned long long* keys   = (unsigned long long*)alloc((size_t)MPAD * 8);
    int*                sel    = (int*)alloc((size_t)NP * sizeof(int));
    float*              canvas = (float*)alloc((size_t)HH * WW * CC * sizeof(float));
    float*              gmax   = (float*)alloc(256);

    pack_mem_kernel<<<NMEM, 256, 0, stream>>>(mem, memh, mnorm2);
    pack_u_kernel<<<MPAD, 256, 0, stream>>>(image, uh, unorm2);
    init_kernel<<<(MPAD + 255) / 256, 256, 0, stream>>>(keys, gmax);

    const int total_tiles = (MPAD / 64) * (NMEM / 64);   // 249 * 64 = 15936
    gemm_argmin_kernel<<<total_tiles / 8, 256, 0, stream>>>(uh, memh, unorm2,
                                                            mnorm2, keys);

    select_kernel<<<(NP + 255) / 256, 256, 0, stream>>>(keys, mapping, sel);
    fold_kernel<<<(HH * WW * CC + 255) / 256, 256, 0, stream>>>(image, mem2, sel,
                                                                canvas, gmax);
    normalize_kernel<<<(HH * WW * CC + 255) / 256, 256, 0, stream>>>(canvas, gmax,
                                                                     out);
}
